// BAKTTime_87840671138362
// MI455X (gfx1250) — compile-verified, matrix operand order checked by reading, not run on hardware
//
#include <hip/hip_runtime.h>
#include <hip/hip_bf16.h>
#include <math.h>

// ---------------------------------------------------------------------------
// CDNA5 (gfx1250): bf16 WMMA GEMMs + flash attention. All WMMA operands are
// packed bf16 with contiguous k so every fragment is two global_load_b128s.
// ---------------------------------------------------------------------------

#define BB 2
#define LL 2048
#define DD 256
#define BLR (BB * LL)
#define NH 8
#define DKH 32
#define DECAYF 0.2f

typedef __attribute__((ext_vector_type(16))) __bf16 v16bf;
typedef __attribute__((ext_vector_type(8)))  float  v8f;

#define DEV __device__ __forceinline__

DEV unsigned short f2bf16u(float f) {            // fp32 -> bf16 bits, RNE
  unsigned u = __builtin_bit_cast(unsigned, f);
  u += 0x7FFFu + ((u >> 16) & 1u);
  return (unsigned short)(u >> 16);
}

DEV v16bf bf_zero() {
  v16bf r;
  unsigned short z = 0;
#pragma unroll
  for (int i = 0; i < 16; ++i) r[i] = __builtin_bit_cast(__bf16, z);
  return r;
}

// Contiguous packed-bf16 fragment load (ISA 7.12.2 16-bit operand layout):
// lane<16 holds k {0..7,16..23}; lane>=16 holds k {8..15,24..31}.
// p must be 16B aligned; lowers to two global_load_b128.
DEV v16bf load_frag16(const unsigned short* __restrict__ p, bool hi) {
  const int b0 = hi ? 8 : 0;
  union { v16bf v; uint4 q[2]; } u;
  u.q[0] = *(const uint4*)(p + b0);
  u.q[1] = *(const uint4*)(p + b0 + 16);
  return u.v;
}

DEV v8f wmma_bf16(v16bf a, v16bf b, v8f c) {
  return __builtin_amdgcn_wmma_f32_16x16x32_bf16(false, a, false, b, (short)0, c,
                                                 false, false);
}

DEV float wave_sum(float v) {
#pragma unroll
  for (int m = 16; m >= 1; m >>= 1) v += __shfl_xor(v, m, 32);
  return v;
}
DEV float half_max(float v) {
#pragma unroll
  for (int m = 8; m >= 1; m >>= 1) v = fmaxf(v, __shfl_xor(v, m, 32));
  return v;
}
DEV float half_sum(float v) {
#pragma unroll
  for (int m = 8; m >= 1; m >>= 1) v += __shfl_xor(v, m, 32);
  return v;
}

// ---------------------------------------------------------------------------
// One-shot converters and tiled bf16 transpose.
// ---------------------------------------------------------------------------
__global__ void k_cvt(const float* __restrict__ in, unsigned short* __restrict__ out,
                      long n) {
  const long i = (long)blockIdx.x * blockDim.x + threadIdx.x;
  if (i < n) out[i] = f2bf16u(in[i]);
}

// conv weight (Dout, Din, K) -> bf16 (K, Dout, Din) so fragments are contiguous
__global__ void k_cvtconv(const float* __restrict__ in, unsigned short* __restrict__ out,
                          int K) {
  const long i = (long)blockIdx.x * blockDim.x + threadIdx.x;
  const long n = (long)DD * DD * K;
  if (i >= n) return;
  const int t = (int)(i % K);
  const long nk = i / K;            // n*DD + k
  out[((long)t * DD + nk / DD) * DD + (nk % DD)] = f2bf16u(in[i]);
}

// per-batch bf16 transpose: in [z][R][C] -> out [z][C][R], 32x32 LDS tiles
__global__ void k_tr(const unsigned short* __restrict__ in,
                     unsigned short* __restrict__ out, int R, int C) {
  __shared__ unsigned short t[32][33];
  const unsigned short* ib = in + (long)blockIdx.z * R * C;
  unsigned short* ob = out + (long)blockIdx.z * R * C;
  const int tx = threadIdx.x & 31, ty = threadIdx.x >> 5;   // 8 rows of 32
  const int r0 = blockIdx.y * 32, c0 = blockIdx.x * 32;
#pragma unroll
  for (int j = 0; j < 4; ++j) {
    const int r = ty + j * 8;
    t[r][tx] = ib[(long)(r0 + r) * C + c0 + tx];
  }
  __syncthreads();
#pragma unroll
  for (int j = 0; j < 4; ++j) {
    const int r = ty + j * 8;
    ob[(long)(c0 + r) * R + r0 + tx] = t[tx][r];
  }
}

// ---------------------------------------------------------------------------
// Kernel 1: dual causal conv (K=3,5) as tap-shifted WMMA GEMMs + combine +
// residual; writes pre-LN activations (fp32). 5 shared A fragments per k-step
// (shifts -4..0); guarded path only for the batch-head tile (l0 == 0).
// ---------------------------------------------------------------------------
template <bool GUARD>
DEV void conv_accum(const unsigned short* __restrict__ xb,
                    const unsigned short* __restrict__ w3b,
                    const unsigned short* __restrict__ w5b,
                    int l0, int ln, int tn, bool hi, v8f& c3, v8f& c5) {
  for (int kb = 0; kb < DD; kb += 32) {
    v16bf a[5];
#pragma unroll
    for (int s = 0; s < 5; ++s) {
      const int lr = l0 + ln + s - 4;
      if (GUARD)
        a[s] = (lr >= 0) ? load_frag16(xb + (long)lr * DD + kb, hi) : bf_zero();
      else
        a[s] = load_frag16(xb + (long)lr * DD + kb, hi);
    }
#pragma unroll
    for (int t = 0; t < 3; ++t) {
      v16bf w = load_frag16(w3b + ((long)t * DD + tn + ln) * DD + kb, hi);
      c3 = wmma_bf16(a[t + 2], w, c3);   // shift t-2 == (t+2)-4
    }
#pragma unroll
    for (int t = 0; t < 5; ++t) {
      v16bf w = load_frag16(w5b + ((long)t * DD + tn + ln) * DD + kb, hi);
      c5 = wmma_bf16(a[t], w, c5);       // shift t-4
    }
  }
}

__global__ void k_freq(const float* __restrict__ x,
                       const unsigned short* __restrict__ xb16,
                       const unsigned short* __restrict__ w3b,
                       const unsigned short* __restrict__ w5b,
                       const float* __restrict__ b3, const float* __restrict__ b5,
                       const float* __restrict__ be3p, const float* __restrict__ be5p,
                       const float* __restrict__ combw,
                       float* __restrict__ pre) {
  const int wave = threadIdx.x >> 5, lane = threadIdx.x & 31;
  const bool hi = lane >= 16;
  const int ln = lane & 15;
  const int tiles_n = DD / 16;
  const int tile = blockIdx.x * 8 + wave;
  const int tm = (tile / tiles_n) * 16;
  const int tn = (tile % tiles_n) * 16;
  const int b = tm / LL, l0 = tm % LL;
  const unsigned short* xb = xb16 + (long)b * LL * DD;

  v8f c3 = {}, c5 = {};
  if (l0 == 0)
    conv_accum<true>(xb, w3b, w5b, l0, ln, tn, hi, c3, c5);
  else
    conv_accum<false>(xb, w3b, w5b, l0, ln, tn, hi, c3, c5);

  const int col = tn + ln;
  const float bias3 = b3[col], bias5 = b5[col];
  float be3 = be3p[col]; be3 *= be3;
  float be5 = be5p[col]; be5 *= be5;
  const float w0 = combw[0], w1 = combw[1];
  const float mx = fmaxf(w0, w1);
  const float e0 = expf(w0 - mx), e1 = expf(w1 - mx);
  const float cw0 = e0 / (e0 + e1), cw1 = e1 / (e0 + e1);
#pragma unroll
  for (int r = 0; r < 8; ++r) {
    const long row = tm + r + (hi ? 8 : 0);
    const float xv = x[row * DD + col];
    const float lo3 = c3[r] + bias3;
    const float lo5 = c5[r] + bias5;
    const float m3 = lo3 + be3 * (xv - lo3);
    const float m5 = lo5 + be5 * (xv - lo5);
    pre[row * DD + col] = cw0 * m3 + cw1 * m5 + xv;
  }
}

// ---------------------------------------------------------------------------
// Kernel 2: LayerNorm, one wave per 256-wide row (fp32 in/out).
// ---------------------------------------------------------------------------
__global__ void k_ln(const float* __restrict__ pre, const float* __restrict__ lw,
                     const float* __restrict__ lb, float* __restrict__ h) {
  const int wave = threadIdx.x >> 5, lane = threadIdx.x & 31;
  const long row = (long)blockIdx.x * 8 + wave;
  const float* p = pre + row * DD;
  float v[8], s = 0.f;
#pragma unroll
  for (int j = 0; j < 8; ++j) { v[j] = p[j * 32 + lane]; s += v[j]; }
  const float mean = wave_sum(s) * (1.0f / DD);
  float qacc = 0.f;
#pragma unroll
  for (int j = 0; j < 8; ++j) { float d = v[j] - mean; qacc += d * d; }
  const float var = wave_sum(qacc) * (1.0f / DD);
  const float inv = 1.0f / sqrtf(var + 1e-12f);
#pragma unroll
  for (int j = 0; j < 8; ++j) {
    const int d = j * 32 + lane;
    h[row * DD + d] = lw[d] * (v[j] - mean) * inv + lb[d];
  }
}

// ---------------------------------------------------------------------------
// Kernel 3: depthwise trends + dynamic routing; writes bf16 capsule outputs.
// ---------------------------------------------------------------------------
DEV void route2(const float u0[8], const float u1[8], float out[8]) {
  float g0 = 0.f, g1 = 0.f;
#pragma unroll
  for (int it = 0; it < 3; ++it) {
    const float mx = fmaxf(g0, g1);
    const float e0 = expf(g0 - mx), e1 = expf(g1 - mx);
    const float c0 = e0 / (e0 + e1), c1 = e1 / (e0 + e1);
    float s[8], n2l = 0.f;
#pragma unroll
    for (int j = 0; j < 8; ++j) { s[j] = c0 * u0[j] + c1 * u1[j]; n2l += s[j] * s[j]; }
    const float n2 = wave_sum(n2l);
    const float sc = (n2 / (1.0f + n2)) / (sqrtf(n2) + 1e-9f);
#pragma unroll
    for (int j = 0; j < 8; ++j) out[j] = sc * s[j];
    if (it < 2) {
      float d0 = 0.f, d1 = 0.f;
#pragma unroll
      for (int j = 0; j < 8; ++j) { d0 += u0[j] * out[j]; d1 += u1[j] * out[j]; }
      g0 += wave_sum(d0);
      g1 += wave_sum(d1);
    }
  }
}

__global__ void k_caps(const float* __restrict__ h, const float* __restrict__ dw3,
                       const float* __restrict__ dw5,
                       unsigned short* __restrict__ srb,
                       unsigned short* __restrict__ trb) {
  const int wave = threadIdx.x >> 5, lane = threadIdx.x & 31;
  const long row = (long)blockIdx.x * 8 + wave;
  const int l = (int)(row % LL);
  float hv[8], t3[8], t5[8];
#pragma unroll
  for (int j = 0; j < 8; ++j) {
    const int d = j * 32 + lane;
    hv[j] = h[row * DD + d];
    float a3 = 0.f;
#pragma unroll
    for (int k = 0; k < 3; ++k) {
      const int ls = l - 2 + k;
      if (ls >= 0) a3 += h[(row - 2 + k) * DD + d] * dw3[d * 3 + k];
    }
    t3[j] = a3;
    float a5 = 0.f;
#pragma unroll
    for (int k = 0; k < 5; ++k) {
      const int ls = l - 4 + k;
      if (ls >= 0) a5 += h[(row - 4 + k) * DD + d] * dw5[d * 5 + k];
    }
    t5[j] = a5;
  }
  float u0[8], u1[8], out[8];
#pragma unroll
  for (int j = 0; j < 8; ++j) { u0[j] = hv[j] - t3[j]; u1[j] = hv[j] - t5[j]; }
  route2(u0, u1, out);
#pragma unroll
  for (int j = 0; j < 8; ++j) srb[row * DD + j * 32 + lane] = f2bf16u(out[j]);
  route2(t3, t5, out);
#pragma unroll
  for (int j = 0; j < 8; ++j) trb[row * DD + j * 32 + lane] = f2bf16u(out[j]);
}

// ---------------------------------------------------------------------------
// Kernel 4: time attention (causal, exp(-DECAY*dist) prior), flash-style.
// X row-major feeds Q/K fragments; XT ([C][L]) feeds P*V fragments, so every
// operand is two b128 loads. One wave per 16-query block.
// ---------------------------------------------------------------------------
__global__ void k_tattn(const unsigned short* __restrict__ srb,
                        const unsigned short* __restrict__ trb,
                        const unsigned short* __restrict__ srbT,
                        const unsigned short* __restrict__ trbT,
                        unsigned short* __restrict__ sfb,
                        unsigned short* __restrict__ tfb) {
  const int wave = threadIdx.x >> 5, lane = threadIdx.x & 31;
  const bool hi = lane >= 16;
  const int ln = lane & 15;
  const int inst = blockIdx.y;
  const long ioff = (long)(inst & 1) * LL * DD;
  const unsigned short* X  = (inst < 2 ? srb : trb) + ioff;
  const unsigned short* XT = (inst < 2 ? srbT : trbT) + ioff;
  unsigned short* O = (inst < 2 ? sfb : tfb) + ioff;
  const int q0 = (blockIdx.x * 8 + wave) * 16;

  __shared__ unsigned short pls[8][16][32];   // bf16 P tile per wave
  unsigned short (*pl)[32] = pls[wave];

  v16bf aq[8];
#pragma unroll
  for (int s = 0; s < 8; ++s) aq[s] = load_frag16(X + (long)(q0 + ln) * DD + s * 32, hi);

  v8f o[16];
  v8f zf = {};
#pragma unroll
  for (int ct = 0; ct < 16; ++ct) o[ct] = zf;
  float mrow[8], lrow[8];
#pragma unroll
  for (int r = 0; r < 8; ++r) { mrow[r] = -1e30f; lrow[r] = 0.f; }

  for (int k0 = 0; k0 < q0 + 16; k0 += 32) {
    const int key0 = k0 + ln, key1 = k0 + 16 + ln;
    v8f s0 = zf, s1 = zf;
#pragma unroll
    for (int s = 0; s < 8; ++s) {
      v16bf b0 = load_frag16(X + (long)key0 * DD + s * 32, hi);
      s0 = wmma_bf16(aq[s], b0, s0);
      v16bf b1 = (key1 < LL) ? load_frag16(X + (long)key1 * DD + s * 32, hi)
                             : bf_zero();
      s1 = wmma_bf16(aq[s], b1, s1);
    }
    float p0[8], p1[8];
#pragma unroll
    for (int r = 0; r < 8; ++r) {
      const int qr = q0 + r + (hi ? 8 : 0);
      const float l0v = (key0 <= qr) ? s0[r] - DECAYF * (float)(qr - key0) : -1e30f;
      const float l1v = (key1 <= qr) ? s1[r] - DECAYF * (float)(qr - key1) : -1e30f;
      const float rm = half_max(fmaxf(l0v, l1v));
      const float mn = fmaxf(mrow[r], rm);
      const float alpha = expf(mrow[r] - mn);
      p0[r] = expf(l0v - mn);
      p1[r] = expf(l1v - mn);
      const float rs = half_sum(p0[r] + p1[r]);
      lrow[r] = lrow[r] * alpha + rs;
      mrow[r] = mn;
#pragma unroll
      for (int ct = 0; ct < 16; ++ct) o[ct][r] *= alpha;
    }
    const int prow = hi ? 8 : 0;
#pragma unroll
    for (int r = 0; r < 8; ++r) {
      pl[prow + r][ln]      = f2bf16u(p0[r]);
      pl[prow + r][16 + ln] = f2bf16u(p1[r]);
    }
    // rebuild P (16x32 bf16) as an A fragment: two ds_load_b128 per lane
    const int b0i = hi ? 8 : 0;
    union { v16bf v; uint4 q[2]; } pu;
    pu.q[0] = *(const uint4*)&pl[ln][b0i];
    pu.q[1] = *(const uint4*)&pl[ln][b0i + 16];
#pragma unroll
    for (int ct = 0; ct < 16; ++ct) {
      v16bf bv = load_frag16(XT + (long)(ct * 16 + ln) * LL + k0, hi);
      o[ct] = wmma_bf16(pu.v, bv, o[ct]);
    }
  }
#pragma unroll
  for (int ct = 0; ct < 16; ++ct) {
#pragma unroll
    for (int r = 0; r < 8; ++r) {
      const long row = q0 + r + (hi ? 8 : 0);
      O[row * DD + ct * 16 + ln] = f2bf16u(o[ct][r] / lrow[r]);
    }
  }
}

// ---------------------------------------------------------------------------
// Kernel 5: GEMM  C[M,N] = [A0|A1][M,K or 2K] * W^T + bias.
// A0/A1, W packed bf16. Output fp32 (Cf) and/or bf16 (Cb).
// ---------------------------------------------------------------------------
__global__ void k_gemm(const unsigned short* __restrict__ A0,
                       const unsigned short* __restrict__ A1, int K, int lda,
                       const unsigned short* __restrict__ W, int ldw,
                       const float* __restrict__ bias,
                       float* __restrict__ Cf, unsigned short* __restrict__ Cb,
                       int ldc, int M, int N) {
  const int wave = threadIdx.x >> 5, lane = threadIdx.x & 31;
  const bool hi = lane >= 16;
  const int ln = lane & 15;
  const int tiles_n = N >> 4;
  const int tile = blockIdx.x * 8 + wave;
  const int tm = (tile / tiles_n) << 4;
  const int tn = (tile % tiles_n) << 4;
  if (tm >= M) return;
  const unsigned short* ap0 = A0 + (long)(tm + ln) * lda;
  const unsigned short* wp = W + (long)(tn + ln) * ldw;
  v8f c = {};
  for (int kb = 0; kb < K; kb += 32)
    c = wmma_bf16(load_frag16(ap0 + kb, hi), load_frag16(wp + kb, hi), c);
  if (A1) {
    const unsigned short* ap1 = A1 + (long)(tm + ln) * lda;
    for (int kb = 0; kb < K; kb += 32)
      c = wmma_bf16(load_frag16(ap1 + kb, hi), load_frag16(wp + K + kb, hi), c);
  }
  const int col = tn + ln;
  const float bv = bias ? bias[col] : 0.f;
#pragma unroll
  for (int r = 0; r < 8; ++r) {
    const long row = tm + r + (hi ? 8 : 0);
    const float acc = c[r] + bv;
    if (Cf) Cf[row * ldc + col] = acc;
    if (Cb) Cb[row * ldc + col] = f2bf16u(acc);
  }
}

// ---------------------------------------------------------------------------
// Kernel 6: causal MHA (H=8, dk=32) flash-style; zero_pad on query row 0.
// bf16 q/k in row-major, v via vT ([B][C][L]); bf16 attn out.
// ---------------------------------------------------------------------------
__global__ void k_mha(const unsigned short* __restrict__ q,
                      const unsigned short* __restrict__ k,
                      const unsigned short* __restrict__ vT,
                      unsigned short* __restrict__ attn) {
  const int wave = threadIdx.x >> 5, lane = threadIdx.x & 31;
  const bool hi = lane >= 16;
  const int ln = lane & 15;
  const int qbpl = LL / 16;
  const int inst = blockIdx.x * 8 + wave;
  const int qb = inst % qbpl;
  const int bh = inst / qbpl;
  const int b = bh / NH, hh = bh % NH;
  const long base = (long)b * LL * DD + hh * DKH;
  const unsigned short* Q = q + base;
  const unsigned short* K = k + base;
  const unsigned short* VT = vT + (long)b * DD * LL + (long)hh * DKH * LL;
  const int q0 = qb * 16;
  const float scale = 0.17677669529663687f;   // 1/sqrt(32)

  __shared__ unsigned short pls[8][16][32];
  unsigned short (*pl)[32] = pls[wave];

  v16bf aq = load_frag16(Q + (long)(q0 + ln) * DD, hi);
  v8f zf = {};
  v8f o0 = zf, o1 = zf;
  float mrow[8], lrow[8];
#pragma unroll
  for (int r = 0; r < 8; ++r) { mrow[r] = -1e30f; lrow[r] = 0.f; }

  for (int k0 = 0; k0 < q0 + 16; k0 += 32) {
    const int key0 = k0 + ln, key1 = k0 + 16 + ln;
    v16bf b0 = load_frag16(K + (long)key0 * DD, hi);
    v8f s0 = wmma_bf16(aq, b0, zf);
    v16bf b1 = (key1 < LL) ? load_frag16(K + (long)key1 * DD, hi) : bf_zero();
    v8f s1 = wmma_bf16(aq, b1, zf);
    float p0[8], p1[8];
#pragma unroll
    for (int r = 0; r < 8; ++r) {
      const int qr = q0 + r + (hi ? 8 : 0);
      const float l0v = (key0 <= qr) ? s0[r] * scale : -1e30f;
      const float l1v = (key1 <= qr) ? s1[r] * scale : -1e30f;
      const float rm = half_max(fmaxf(l0v, l1v));
      const float mn = fmaxf(mrow[r], rm);
      const float alpha = expf(mrow[r] - mn);
      p0[r] = expf(l0v - mn);
      p1[r] = expf(l1v - mn);
      const float rs = half_sum(p0[r] + p1[r]);
      lrow[r] = lrow[r] * alpha + rs;
      mrow[r] = mn;
      o0[r] *= alpha;
      o1[r] *= alpha;
    }
    const int prow = hi ? 8 : 0;
#pragma unroll
    for (int r = 0; r < 8; ++r) {
      pl[prow + r][ln]      = f2bf16u(p0[r]);
      pl[prow + r][16 + ln] = f2bf16u(p1[r]);
    }
    const int b0i = hi ? 8 : 0;
    union { v16bf v; uint4 q[2]; } pu;
    pu.q[0] = *(const uint4*)&pl[ln][b0i];
    pu.q[1] = *(const uint4*)&pl[ln][b0i + 16];
    o0 = wmma_bf16(pu.v, load_frag16(VT + (long)ln * LL + k0, hi), o0);
    o1 = wmma_bf16(pu.v, load_frag16(VT + (long)(16 + ln) * LL + k0, hi), o1);
  }
#pragma unroll
  for (int r = 0; r < 8; ++r) {
    const int l = q0 + r + (hi ? 8 : 0);
    const float inv = (l == 0) ? 0.f : 1.0f / lrow[r];   // zero_pad row 0
    attn[((long)b * LL + l) * DD + hh * DKH + ln]      = f2bf16u(o0[r] * inv);
    attn[((long)b * LL + l) * DD + hh * DKH + 16 + ln] = f2bf16u(o1[r] * inv);
  }
}

// ---------------------------------------------------------------------------
extern "C" void kernel_launch(void* const* d_in, const int* in_sizes, int n_in,
                              void* d_out, int out_size, void* d_ws, size_t ws_size,
                              hipStream_t stream) {
  const float* x   = (const float*)d_in[0];
  const float* w3  = (const float*)d_in[1];
  const float* b3  = (const float*)d_in[2];
  const float* w5  = (const float*)d_in[3];
  const float* b5  = (const float*)d_in[4];
  const float* be3 = (const float*)d_in[5];
  const float* be5 = (const float*)d_in[6];
  const float* cw  = (const float*)d_in[7];
  const float* lnw = (const float*)d_in[8];
  const float* lnb = (const float*)d_in[9];
  const float* dw3 = (const float*)d_in[10];
  const float* dw5 = (const float*)d_in[11];
  const float* fw  = (const float*)d_in[12];
  const float* fb  = (const float*)d_in[13];
  const float* wq  = (const float*)d_in[14];
  const float* bq  = (const float*)d_in[15];
  const float* wk  = (const float*)d_in[16];
  const float* bk  = (const float*)d_in[17];
  const float* wv  = (const float*)d_in[18];
  const float* bv  = (const float*)d_in[19];
  const float* wo  = (const float*)d_in[20];
  const float* bo  = (const float*)d_in[21];

  const long S = (long)BLR * DD;          // 1,048,576 elements
  char* w = (char*)d_ws;
  float* pre = (float*)w;               w += S * 4;
  float* h   = (float*)w;               w += S * 4;
  unsigned short* xb16  = (unsigned short*)w; w += S * 2;
  unsigned short* srb   = (unsigned short*)w; w += S * 2;
  unsigned short* trb   = (unsigned short*)w; w += S * 2;
  unsigned short* srbT  = (unsigned short*)w; w += S * 2;
  unsigned short* trbT  = (unsigned short*)w; w += S * 2;
  unsigned short* sfb   = (unsigned short*)w; w += S * 2;
  unsigned short* tfb   = (unsigned short*)w; w += S * 2;
  unsigned short* fusedb= (unsigned short*)w; w += S * 2;
  unsigned short* qb16  = (unsigned short*)w; w += S * 2;
  unsigned short* kb16  = (unsigned short*)w; w += S * 2;
  unsigned short* vb16  = (unsigned short*)w; w += S * 2;
  unsigned short* vbT   = (unsigned short*)w; w += S * 2;
  unsigned short* attnb = (unsigned short*)w; w += S * 2;
  unsigned short* w3b   = (unsigned short*)w; w += (long)3 * DD * DD * 2;
  unsigned short* w5b   = (unsigned short*)w; w += (long)5 * DD * DD * 2;
  unsigned short* fwb   = (unsigned short*)w; w += (long)DD * 2 * DD * 2;
  unsigned short* wqb   = (unsigned short*)w; w += (long)DD * DD * 2;
  unsigned short* wkb   = (unsigned short*)w; w += (long)DD * DD * 2;
  unsigned short* wvb   = (unsigned short*)w; w += (long)DD * DD * 2;
  unsigned short* wob   = (unsigned short*)w; w += (long)DD * DD * 2;

  const dim3 blk(256);
  const int gemm_blocks = (BLR / 16) * (DD / 16) / 8;   // 512
  const dim3 trgrid(DD / 32, LL / 32, BB);              // (8, 64, 2)

  // one-shot bf16 quantization of inputs/weights
  k_cvt<<<dim3((int)((S + 255) / 256)), blk, 0, stream>>>(x, xb16, S);
  k_cvtconv<<<dim3((3 * DD * DD + 255) / 256), blk, 0, stream>>>(w3, w3b, 3);
  k_cvtconv<<<dim3((5 * DD * DD + 255) / 256), blk, 0, stream>>>(w5, w5b, 5);
  k_cvt<<<dim3((DD * 2 * DD + 255) / 256), blk, 0, stream>>>(fw, fwb, DD * 2 * DD);
  k_cvt<<<dim3((DD * DD + 255) / 256), blk, 0, stream>>>(wq, wqb, DD * DD);
  k_cvt<<<dim3((DD * DD + 255) / 256), blk, 0, stream>>>(wk, wkb, DD * DD);
  k_cvt<<<dim3((DD * DD + 255) / 256), blk, 0, stream>>>(wv, wvb, DD * DD);
  k_cvt<<<dim3((DD * DD + 255) / 256), blk, 0, stream>>>(wo, wob, DD * DD);

  k_freq<<<dim3(gemm_blocks), blk, 0, stream>>>(x, xb16, w3b, w5b, b3, b5, be3, be5,
                                                cw, pre);
  k_ln<<<dim3(BLR / 8), blk, 0, stream>>>(pre, lnw, lnb, h);
  k_caps<<<dim3(BLR / 8), blk, 0, stream>>>(h, dw3, dw5, srb, trb);
  k_tr<<<trgrid, blk, 0, stream>>>(srb, srbT, LL, DD);
  k_tr<<<trgrid, blk, 0, stream>>>(trb, trbT, LL, DD);
  k_tattn<<<dim3(LL / 16 / 8, 4), blk, 0, stream>>>(srb, trb, srbT, trbT, sfb, tfb);

  // fused = [sf|tf] @ fusion_w^T + fusion_b  (dual-A concat GEMM, bf16 out)
  k_gemm<<<dim3(gemm_blocks), blk, 0, stream>>>(sfb, tfb, DD, DD, fwb, 2 * DD, fb,
                                                nullptr, fusedb, DD, BLR, DD);
  k_gemm<<<dim3(gemm_blocks), blk, 0, stream>>>(fusedb, nullptr, DD, DD, wqb, DD, bq,
                                                nullptr, qb16, DD, BLR, DD);
  k_gemm<<<dim3(gemm_blocks), blk, 0, stream>>>(fusedb, nullptr, DD, DD, wkb, DD, bk,
                                                nullptr, kb16, DD, BLR, DD);
  k_gemm<<<dim3(gemm_blocks), blk, 0, stream>>>(fusedb, nullptr, DD, DD, wvb, DD, bv,
                                                nullptr, vb16, DD, BLR, DD);
  k_tr<<<trgrid, blk, 0, stream>>>(vb16, vbT, LL, DD);

  k_mha<<<dim3(BB * NH * (LL / 16) / 8), blk, 0, stream>>>(qb16, kb16, vbT, attnb);

  k_gemm<<<dim3(gemm_blocks), blk, 0, stream>>>(attnb, nullptr, DD, DD, wob, DD, bo,
                                                (float*)d_out, nullptr, DD, BLR, DD);
}
